// ConcatConvLayer_28149215658670
// MI455X (gfx1250) — matrix operand order repaired and checked by reading, not compiled, not run on hardware
//
#include <hip/hip_runtime.h>
#include <hip/hip_bf16.h>
#include <cstdint>
#include <cstddef>

typedef __bf16 bf16;
typedef __attribute__((ext_vector_type(16))) __bf16 v16bf;
typedef __attribute__((ext_vector_type(8)))  __bf16 v8bf;
typedef __attribute__((ext_vector_type(4)))  __bf16 v4bf;
typedef __attribute__((ext_vector_type(8)))  float  v8f;
typedef int v4i __attribute__((vector_size(16)));   // matches builtin pointee type

#define NODE_DIM 128
#define EDGE_DIM 64
#define MNBR     16
#define NPB      50   // nodes per block in the main kernel

// ---------------------------------------------------------------------------
// CDNA5 async-to-LDS helpers (ASYNCcnt-tracked DMA of 16B per lane).
// ---------------------------------------------------------------------------
__device__ __forceinline__ void async_copy16(const void* g, void* l) {
#if __has_builtin(__builtin_amdgcn_global_load_async_to_lds_b128)
  __builtin_amdgcn_global_load_async_to_lds_b128(
      (__attribute__((address_space(1))) v4i*)(uintptr_t)g,
      (__attribute__((address_space(3))) v4i*)(uint32_t)(uintptr_t)l,
      0, 0);
#else
  uint32_t loff = (uint32_t)(uintptr_t)l;   // low 32 bits of generic = LDS addr
  uint64_t ga   = (uint64_t)(uintptr_t)g;
  asm volatile("global_load_async_to_lds_b128 %0, %1, off"
               :: "v"(loff), "v"(ga) : "memory");
#endif
}

template <int Ncnt>
__device__ __forceinline__ void wait_asynccnt() {
#if __has_builtin(__builtin_amdgcn_s_wait_asynccnt)
  __builtin_amdgcn_s_wait_asynccnt(Ncnt);
#else
  asm volatile("s_wait_asynccnt %0" :: "i"(Ncnt) : "memory");
#endif
}

// ---------------------------------------------------------------------------
// WMMA fragment helpers
// ---------------------------------------------------------------------------
// 16x32 bf16 A-fragment (CDNA5 layout): lanes 0-15 row M=lane, K=k0..k0+7 and
// k0+16..k0+23 with k0=0; lanes 16-31 same rows with k0=8. p -> (row, k0).
__device__ __forceinline__ v16bf ldA16(const bf16* p) {
  v8bf lo = *(const v8bf*)p;
  v8bf hi = *(const v8bf*)(p + 16);
  v16bf a;
#pragma unroll
  for (int i = 0; i < 8; ++i) { a[i] = lo[i]; a[i + 8] = hi[i]; }
  return a;
}

// Same A-fragment but sourced from fp32 LDS data (converted to bf16).
__device__ __forceinline__ v16bf ldA16f(const float* p) {
  v16bf a;
#pragma unroll
  for (int i = 0; i < 8; ++i) { a[i] = (bf16)p[i]; a[i + 8] = (bf16)p[i + 16]; }
  return a;
}

// 32x16 bf16 B-fragment from row-major fp32 W[K x ldw]:
// lanes 0-15 col N=lane, K=0..7,16..23; lanes 16-31 K=8..15,24..31.
__device__ __forceinline__ v16bf ldB_global(const float* __restrict__ W, int ldw,
                                            int kc, int ncol, int lane) {
  const int n  = ncol + (lane & 15);
  const int k0 = kc * 32 + ((lane & 16) ? 8 : 0);
  v16bf b;
#pragma unroll
  for (int i = 0; i < 16; ++i) b[i] = (bf16)W[(size_t)(k0 + i + (i >= 8 ? 8 : 0)) * ldw + n];
  return b;
}

// ---------------------------------------------------------------------------
// Pass 1: LayerNorm(x) -> bf16 x_normed in workspace. One wave32 per row.
// ---------------------------------------------------------------------------
__global__ __launch_bounds__(256) void cgcnn_ln(
    const float* __restrict__ x, const float* __restrict__ ln_scale,
    const float* __restrict__ ln_bias, bf16* __restrict__ xn, int N)
{
  const int wave = threadIdx.x >> 5;
  const int lane = threadIdx.x & 31;
  const int row  = blockIdx.x * 8 + wave;
  if (row >= N) return;

  const int c = lane * 4;
  const float4 v = *(const float4*)&x[(size_t)row * NODE_DIM + c];

  float s = v.x + v.y + v.z + v.w;
#pragma unroll
  for (int off = 16; off > 0; off >>= 1) s += __shfl_xor(s, off, 32);
  const float mu = s * (1.0f / 128.0f);

  const float dx = v.x - mu, dy = v.y - mu, dz = v.z - mu, dw = v.w - mu;
  float q = dx * dx + dy * dy + dz * dz + dw * dw;
#pragma unroll
  for (int off = 16; off > 0; off >>= 1) q += __shfl_xor(q, off, 32);
  const float inv = rsqrtf(q * (1.0f / 128.0f) + 1e-6f);

  const float4 sc = *(const float4*)&ln_scale[c];
  const float4 bi = *(const float4*)&ln_bias[c];
  v4bf o;
  o[0] = (bf16)(dx * inv * sc.x + bi.x);
  o[1] = (bf16)(dy * inv * sc.y + bi.y);
  o[2] = (bf16)(dz * inv * sc.z + bi.z);
  o[3] = (bf16)(dw * inv * sc.w + bi.w);
  *(v4bf*)&xn[(size_t)row * NODE_DIM + c] = o;
}

// ---------------------------------------------------------------------------
// Pass 2: per node: concat[16x320] @ W1 -> SiLU -> @ W2 -> row-sum + residual.
// 8 waves/block, wave j owns output columns [16j, 16j+16). Weight B-fragments
// are register-resident bf16. Node staging is double-buffered in LDS and
// filled with global_load_async_to_lds_b128 (ASYNCcnt) one node ahead.
// ---------------------------------------------------------------------------
__global__ __launch_bounds__(256) void cgcnn_main(
    const float* __restrict__ x, const float* __restrict__ nbr_fea,
    const int* __restrict__ nbr_idx,
    const float* __restrict__ W1, const float* __restrict__ b1,
    const float* __restrict__ W2, const float* __restrict__ b2,
    const bf16* __restrict__ xn, float* __restrict__ out, int N)
{
  __shared__ __align__(16) bf16  sCb[2][NODE_DIM];          // center row (bf16)
  __shared__ __align__(16) bf16  sNb[2][MNBR * NODE_DIM];   // gathered rows
  __shared__ __align__(16) float sEf[2][MNBR * EDGE_DIM];   // edges (fp32)
  __shared__ __align__(16) bf16  sH[MNBR * NODE_DIM];       // SiLU(h)

  const int tid  = threadIdx.x;
  const int wave = tid >> 5;
  const int lane = tid & 31;

  const int base = blockIdx.x * NPB;
  if (base >= N) return;

  // Register-resident B fragments (bf16), loaded once per block.
  v16bf w1f[10];
#pragma unroll
  for (int kc = 0; kc < 10; ++kc) w1f[kc] = ldB_global(W1, NODE_DIM, kc, wave * 16, lane);
  v16bf w2f[4];
#pragma unroll
  for (int kc = 0; kc < 4; ++kc)  w2f[kc] = ldB_global(W2, NODE_DIM, kc, wave * 16, lane);

  const int   col   = wave * 16 + (lane & 15);
  const float bias1 = b1[col];
  const float bias2 = b2[col];

  // Each thread issues exactly 3 async B128 loads per node (uniform ASYNCcnt):
  //  - 1/256 of the neighbor gather (16B of bf16 rows)
  //  - 1/256 of the fp32 edge block (16B)
  //  - 1/16 of the center row, issued redundantly by all 16 thread-groups
  auto issue_node = [&](int n, int b) {
    const int row = tid >> 4, c = (tid & 15) * 8;
    const int nb  = nbr_idx[n * MNBR + row];
    async_copy16(&xn[(size_t)nb * NODE_DIM + c], &sNb[b][row * NODE_DIM + c]);
    async_copy16(&nbr_fea[(size_t)n * MNBR * EDGE_DIM + tid * 4], &sEf[b][tid * 4]);
    async_copy16(&xn[(size_t)n * NODE_DIM + (tid & 15) * 8], &sCb[b][(tid & 15) * 8]);
  };

  issue_node(base, 0);   // prime the pipeline

  for (int i = 0; i < NPB; ++i) {
    const int n = base + i;          // uniform across the block
    if (n >= N) break;
    const int  cur     = i & 1;
    const int  nxt     = cur ^ 1;
    const bool hasNext = (i + 1 < NPB) && (n + 1 < N);

    if (hasNext) { issue_node(n + 1, nxt); wait_asynccnt<3>(); }
    else         { wait_asynccnt<0>(); }
    __syncthreads();   // current buffer complete & visible to all waves

    // ---- GEMM1: concat[16x320] x W1-tile -> [16x16] f32 ----
    const int m   = lane & 15;
    const int off = (lane & 16) ? 8 : 0;
    v16bf af[10];
#pragma unroll
    for (int kc = 0; kc < 4; ++kc)
      af[kc] = ldA16(&sCb[cur][kc * 32 + off]);                         // center (bcast)
#pragma unroll
    for (int kc = 4; kc < 8; ++kc)
      af[kc] = ldA16(&sNb[cur][m * NODE_DIM + (kc * 32 + off - 128)]);  // neighbor
#pragma unroll
    for (int kc = 8; kc < 10; ++kc)
      af[kc] = ldA16f(&sEf[cur][m * EDGE_DIM + (kc * 32 + off - 256)]); // edge (cvt)

    v8f acc = {};
#pragma unroll
    for (int kc = 0; kc < 10; ++kc)
      acc = __builtin_amdgcn_wmma_f32_16x16x32_bf16(
          false, af[kc], false, w1f[kc], (short)0, acc, false, false);

    // ---- bias + SiLU -> sH (bf16) ----
    {
      const int mbase = (lane & 16) ? 8 : 0;
      const int c     = lane & 15;
#pragma unroll
      for (int r = 0; r < 8; ++r) {
        const float v = acc[r] + bias1;
        const float s = v * (1.0f / (1.0f + __expf(-v)));
        sH[(mbase + r) * NODE_DIM + wave * 16 + c] = (bf16)s;
      }
    }
    __syncthreads();

    // ---- GEMM2: h[16x128] x W2-tile -> [16x16] f32 ----
    v16bf hf[4];
#pragma unroll
    for (int kc = 0; kc < 4; ++kc)
      hf[kc] = ldA16(&sH[m * NODE_DIM + kc * 32 + off]);

    v8f acc2 = {};
#pragma unroll
    for (int kc = 0; kc < 4; ++kc)
      acc2 = __builtin_amdgcn_wmma_f32_16x16x32_bf16(
          false, hf[kc], false, w2f[kc], (short)0, acc2, false, false);

    // ---- reduce over the 16 message rows, add 16*b2 and residual ----
    {
      float part = 0.0f;
#pragma unroll
      for (int r = 0; r < 8; ++r) part += acc2[r];   // rows 0-7 or 8-15
      part += __shfl_xor(part, 16, 32);              // combine both halves
      const float total = part + 16.0f * bias2;
      if (lane < 16)
        out[(size_t)n * NODE_DIM + col] =
            x[(size_t)n * NODE_DIM + col] + total;
    }
    __syncthreads();   // protect sH and staging buffers before next iteration
  }
}

// ---------------------------------------------------------------------------
extern "C" void kernel_launch(void* const* d_in, const int* in_sizes, int n_in,
                              void* d_out, int out_size, void* d_ws, size_t ws_size,
                              hipStream_t stream) {
  const float* x        = (const float*)d_in[0];
  const float* nbr_fea  = (const float*)d_in[1];
  const int*   nbr_idx  = (const int*)d_in[2];
  const float* ln_scale = (const float*)d_in[3];
  const float* ln_bias  = (const float*)d_in[4];
  const float* W1       = (const float*)d_in[5];
  const float* b1       = (const float*)d_in[6];
  const float* W2       = (const float*)d_in[7];
  const float* b2       = (const float*)d_in[8];
  float*       out      = (float*)d_out;
  bf16*        xn       = (bf16*)d_ws;   // N*128 bf16 = 12.8 MB

  const int N = in_sizes[0] / NODE_DIM;

  const int lnBlocks = (N + 7) / 8;
  cgcnn_ln<<<lnBlocks, 256, 0, stream>>>(x, ln_scale, ln_bias, xn, N);

  const int mainBlocks = (N + NPB - 1) / NPB;
  cgcnn_main<<<mainBlocks, 256, 0, stream>>>(x, nbr_fea, nbr_idx, W1, b1, W2, b2,
                                             xn, out, N);
}
